// GCN_63788854280593
// MI455X (gfx1250) — compile-verified, matrix operand order checked by reading, not run on hardware
//
#include <hip/hip_runtime.h>
#include <hip/hip_bf16.h>
#include <math.h>

// ---------------------------------------------------------------------------
// CDNA5 (gfx1250, wave32) GCN: SpMM scatter (global_atomic_add_f32) +
// WMMA bf16 GEMM (v_wmma_f32_16x16x32_bf16) + wave L2-normalize.
// ---------------------------------------------------------------------------

typedef __attribute__((ext_vector_type(8)))  __bf16 v8bf;
typedef __attribute__((ext_vector_type(16))) __bf16 v16bf;
typedef __attribute__((ext_vector_type(8)))  float  v8f;

#define N_NODES 100000
#define IN_C    128
#define HID_C   256
#define OUT_C   128

// ------------------------- fp32 -> bf16 convert ----------------------------
__global__ void cvt_f32_bf16(const float* __restrict__ in, __bf16* __restrict__ out, int n) {
    int i = blockIdx.x * blockDim.x + threadIdx.x;
    if (i < n) out[i] = (__bf16)in[i];
}

// ------------------------- SpMM scatter-add --------------------------------
// One wave per edge. Lane L covers channels i*128 + L*4 .. +3 (fully
// coalesced 512B per iteration). Atomic f32 adds, no return (STOREcnt path).
template <int C>
__global__ void spmm_scatter(const float* __restrict__ h,
                             const int*  __restrict__ erow,
                             const int*  __restrict__ ecol,
                             const float* __restrict__ eval,
                             float* __restrict__ out, int E) {
    const int wave = blockIdx.x * (blockDim.x >> 5) + (threadIdx.x >> 5);
    const int lane = threadIdx.x & 31;
    if (wave >= E) return;
    const int   r = erow[wave];
    const int   c = ecol[wave];
    const float v = eval[wave];
    const float* __restrict__ src = h   + (size_t)c * C;
    float*       __restrict__ dst = out + (size_t)r * C;
#pragma unroll
    for (int i = 0; i < C / 128; ++i) {
        const int off = i * 128 + lane * 4;
        const float4 t = *(const float4*)(src + off);
        unsafeAtomicAdd(dst + off + 0, v * t.x);
        unsafeAtomicAdd(dst + off + 1, v * t.y);
        unsafeAtomicAdd(dst + off + 2, v * t.z);
        unsafeAtomicAdd(dst + off + 3, v * t.w);
    }
}

// ------------------------- WMMA GEMM + bias (+ReLU) ------------------------
// out[m][n] = sum_k A[m][k] * W[n][k] + b[n]      (A fp32 -> bf16, W bf16)
// Block: 256 threads = 8 waves; 16-node tile per block staged in LDS as bf16.
// Each wave produces 16x16 output tiles with V_WMMA_F32_16X16X32_BF16.
//
// Fragment layouts (ISA 7.12.2, wave32):
//   A 16x32 bf16 : lane L holds row M=L%16;  elem j -> K = (j<8?0:16) + (L/16)*8 + j%8
//   B 32x16 bf16 : lane L holds col N=L%16;  same K mapping (B = W^T, so each
//                  lane streams row W[n] contiguously -> two b128 loads/kstep)
//   C/D 16x16 f32: lane L, vgpr r -> M = (L/16)*8 + r, N = L%16
template <int CIN, int COUT, bool RELU>
__global__ void gemm_wmma_bias(const float* __restrict__ A,
                               const __bf16* __restrict__ W,
                               const float* __restrict__ bias,
                               float* __restrict__ out) {
    constexpr int PAD = 8;                       // 16B row padding, keeps 16B alignment
    __shared__ __align__(16) __bf16 As[16][CIN + PAD];

    const int node0 = blockIdx.x * 16;
    const int tid   = threadIdx.x;

    // Stage + convert the 16 x CIN fp32 tile into LDS bf16 (coalesced).
    for (int i = tid; i < 16 * CIN; i += 256) {
        const int r = i / CIN, k = i % CIN;
        As[r][k] = (__bf16)A[(size_t)(node0 + r) * CIN + k];
    }
    __syncthreads();

    const int lane = tid & 31;
    const int wave = tid >> 5;
    const int m    = lane & 15;     // A row / B col / D col index
    const int g    = lane >> 4;     // K-half selector

    for (int nt = wave; nt < COUT / 16; nt += 8) {
        const int n = nt * 16 + m;
        const __bf16* __restrict__ wrow = W + (size_t)n * CIN;
        v8f acc = {};
#pragma unroll
        for (int kb = 0; kb < CIN; kb += 32) {
            const v8bf alo = *(const v8bf*)&As[m][kb + g * 8];
            const v8bf ahi = *(const v8bf*)&As[m][kb + 16 + g * 8];
            const v8bf blo = *(const v8bf*)(wrow + kb + g * 8);
            const v8bf bhi = *(const v8bf*)(wrow + kb + 16 + g * 8);
            const v16bf a = __builtin_shufflevector(alo, ahi,
                0,1,2,3,4,5,6,7,8,9,10,11,12,13,14,15);
            const v16bf b = __builtin_shufflevector(blo, bhi,
                0,1,2,3,4,5,6,7,8,9,10,11,12,13,14,15);
            acc = __builtin_amdgcn_wmma_f32_16x16x32_bf16(
                /*neg_a=*/false, a, /*neg_b=*/false, b,
                /*c_mod=*/(short)0, acc, /*reuse_a=*/false, /*reuse_b=*/false);
        }
        const float bv = bias[n];
#pragma unroll
        for (int r = 0; r < 8; ++r) {
            float v = acc[r] + bv;
            if (RELU) v = fmaxf(v, 0.0f);
            out[(size_t)(node0 + g * 8 + r) * COUT + n] = v;
        }
    }
}

// ------------------------- row-wise L2 normalize ---------------------------
// One wave (32 lanes) per node, 128 channels -> float4 per lane.
__global__ void l2_normalize(const float* __restrict__ h, float* __restrict__ out, int n) {
    const int node = blockIdx.x * (blockDim.x >> 5) + (threadIdx.x >> 5);
    const int lane = threadIdx.x & 31;
    if (node >= n) return;
    const float4 t = *(const float4*)(h + (size_t)node * 128 + lane * 4);
    float s = t.x * t.x + t.y * t.y + t.z * t.z + t.w * t.w;
#pragma unroll
    for (int msk = 16; msk; msk >>= 1) s += __shfl_xor(s, msk, 32);
    const float inv = 1.0f / fmaxf(sqrtf(s), 1e-12f);
    float4 o;
    o.x = t.x * inv; o.y = t.y * inv; o.z = t.z * inv; o.w = t.w * inv;
    *(float4*)(out + (size_t)node * 128 + lane * 4) = o;
}

// ---------------------------------------------------------------------------
extern "C" void kernel_launch(void* const* d_in, const int* in_sizes, int n_in,
                              void* d_out, int out_size, void* d_ws, size_t ws_size,
                              hipStream_t stream) {
    const float* x    = (const float*)d_in[0];
    const int*   erow = (const int*)  d_in[1];
    const int*   ecol = (const int*)  d_in[2];
    const float* eval = (const float*)d_in[3];
    const float* W1   = (const float*)d_in[4];
    const float* b1   = (const float*)d_in[5];
    const float* W2   = (const float*)d_in[6];
    const float* b2   = (const float*)d_in[7];
    const float* W3   = (const float*)d_in[8];
    const float* b3   = (const float*)d_in[9];

    const int E = in_sizes[1];
    const int N = in_sizes[0] / IN_C;       // 100000 (multiple of 16)

    // Workspace layout
    char* base = (char*)d_ws;
    const size_t szAgg = (size_t)N * HID_C * sizeof(float);   // 102.4 MB
    const size_t szH   = (size_t)N * HID_C * sizeof(float);   // 102.4 MB
    const size_t szC   = (size_t)N * OUT_C * sizeof(float);   //  51.2 MB
    float*  agg = (float*)(base);
    float*  hB  = (float*)(base + szAgg);
    float*  hC  = (float*)(base + szAgg + szH);
    __bf16* W1b = (__bf16*)(base + szAgg + szH + szC);
    __bf16* W2b = W1b + (size_t)HID_C * IN_C;
    __bf16* W3b = W2b + (size_t)HID_C * HID_C;

    // 0) Weights -> bf16 (cheap, re-done every call for determinism).
    {
        const int n1 = HID_C * IN_C, n2 = HID_C * HID_C, n3 = OUT_C * HID_C;
        cvt_f32_bf16<<<(n1 + 255) / 256, 256, 0, stream>>>(W1, W1b, n1);
        cvt_f32_bf16<<<(n2 + 255) / 256, 256, 0, stream>>>(W2, W2b, n2);
        cvt_f32_bf16<<<(n3 + 255) / 256, 256, 0, stream>>>(W3, W3b, n3);
    }

    const int spmmBlocks = (E + 7) / 8;            // 8 waves/block, 1 edge/wave
    const int gemmBlocks = N / 16;                 // 16-node tiles
    const int normBlocks = (N + 7) / 8;

    // Layer 1: agg = spmm(x); hB = relu(agg @ W1^T + b1)
    hipMemsetAsync(agg, 0, (size_t)N * IN_C * sizeof(float), stream);
    spmm_scatter<IN_C><<<spmmBlocks, 256, 0, stream>>>(x, erow, ecol, eval, agg, E);
    gemm_wmma_bias<IN_C, HID_C, true><<<gemmBlocks, 256, 0, stream>>>(agg, W1b, b1, hB);

    // Layer 2: agg = spmm(hB); hB = relu(agg @ W2^T + b2)
    hipMemsetAsync(agg, 0, szAgg, stream);
    spmm_scatter<HID_C><<<spmmBlocks, 256, 0, stream>>>(hB, erow, ecol, eval, agg, E);
    gemm_wmma_bias<HID_C, HID_C, true><<<gemmBlocks, 256, 0, stream>>>(agg, W2b, b2, hB);

    // Layer 3: agg = spmm(hB); hC = agg @ W3^T + b3
    hipMemsetAsync(agg, 0, szAgg, stream);
    spmm_scatter<HID_C><<<spmmBlocks, 256, 0, stream>>>(hB, erow, ecol, eval, agg, E);
    gemm_wmma_bias<HID_C, OUT_C, false><<<gemmBlocks, 256, 0, stream>>>(agg, W3b, b3, hC);

    // Final L2 normalize -> d_out
    l2_normalize<<<normBlocks, 256, 0, stream>>>(hC, (float*)d_out, N);
}
